// PointTransformerLayer_60507499266137
// MI455X (gfx1250) — matrix operand order repaired and checked by reading, not compiled, not run on hardware
//
#include <hip/hip_runtime.h>
#include <stdint.h>

// ---------------- problem constants ----------------
#define BQ   4
#define MQ   8192
#define DQ   256
#define KNB  16
#define PEH  64
#define ATH  128
#define NPTS (BQ*MQ)

typedef __attribute__((ext_vector_type(16))) __bf16 v16bf;
typedef __attribute__((ext_vector_type(8)))  __bf16 v8bf;
typedef __attribute__((ext_vector_type(8)))  float  v8f;
typedef __attribute__((ext_vector_type(4)))  float  v4f;

// round-half-up fp32->bf16, two at a time, packed with one v_perm_b32
static __device__ inline uint32_t pk2(float a, float b) {
  union { float f; uint32_t u; } x, y; x.f = a; y.f = b;
  // bytes: [x.hi16 | y.hi16]  (little-endian: lo half = x, hi half = y)
  return __builtin_amdgcn_perm(y.u + 0x8000u, x.u + 0x8000u, 0x07060302u);
}
static __device__ inline uint16_t bfbits(float f) {
  union { float f; uint32_t u; } a; a.f = f;
  return (uint16_t)((a.u + 0x8000u) >> 16);
}
static __device__ inline v8f wmma_bf16(v16bf a, v16bf b, v8f c) {
  return __builtin_amdgcn_wmma_f32_16x16x32_bf16(false, a, false, b, (short)0, c,
                                                 false, false);
}
// A-fragment (16x32 bf16) from 32 consecutive fp32 channels of one row.
// lane<16: K = {0..15}+base ; lane>=16: K = {16..31}+base.
static __device__ inline v16bf packA(const float* p0 /*base+hoffA*/,
                                     const float* p1 /*base+8+hoffA*/) {
  v4f f0 = *(const v4f*)p0, f1 = *(const v4f*)(p0 + 4);
  v4f f2 = *(const v4f*)p1, f3 = *(const v4f*)(p1 + 4);
  union { v16bf v; uint32_t u[8]; } r;
  r.u[0] = pk2(f0[0], f0[1]); r.u[1] = pk2(f0[2], f0[3]);
  r.u[2] = pk2(f1[0], f1[1]); r.u[3] = pk2(f1[2], f1[3]);
  r.u[4] = pk2(f2[0], f2[1]); r.u[5] = pk2(f2[2], f2[3]);
  r.u[6] = pk2(f3[0], f3[1]); r.u[7] = pk2(f3[2], f3[3]);
  return r.v;
}
// Re-load a 16x32 bf16 tile from per-wave LDS scratch as an A-fragment.
static __device__ inline v16bf ldsA(const uint16_t* scr, int row, int hoffA) {
  v8bf lo = *(const v8bf*)(scr + row * 32 + hoffA);
  v8bf hi = *(const v8bf*)(scr + row * 32 + 8 + hoffA);
  v16bf a;
#pragma unroll
  for (int j = 0; j < 8; ++j) { a[j] = lo[j]; a[8 + j] = hi[j]; }
  return a;
}

// ---------------- weight fp32 -> bf16 prep ----------------
__global__ void cvt_bf16_kernel(const float* __restrict__ src,
                                uint16_t* __restrict__ dst, int n) {
  int i = blockIdx.x * 256 + threadIdx.x;
  if (i < n) dst[i] = bfbits(src[i]);
}

// ---------------- KNN (top-16, fp32 VALU) ----------------
#define KCH 1024
__global__ void __launch_bounds__(256) knn_kernel(const float* __restrict__ xyz,
                                                  int* __restrict__ idx) {
  __shared__ v4f keys[KCH];   // x,y,z,|k|^2
  const int b = blockIdx.y;
  const int q = blockIdx.x * 256 + threadIdx.x;
  const float qx = xyz[(size_t)(b * MQ + q) * 3 + 0];
  const float qy = xyz[(size_t)(b * MQ + q) * 3 + 1];
  const float qz = xyz[(size_t)(b * MQ + q) * 3 + 2];
  const float qs = qx * qx + qy * qy + qz * qz;
  float best[KNB]; int bidx[KNB];
#pragma unroll
  for (int j = 0; j < KNB; ++j) { best[j] = 3.4e38f; bidx[j] = 0; }
  for (int c0 = 0; c0 < MQ; c0 += KCH) {
    __syncthreads();
    for (int i = threadIdx.x; i < KCH; i += 256) {
      int kk = c0 + i;
      float kx = xyz[(size_t)(b * MQ + kk) * 3 + 0];
      float ky = xyz[(size_t)(b * MQ + kk) * 3 + 1];
      float kz = xyz[(size_t)(b * MQ + kk) * 3 + 2];
      v4f t; t[0] = kx; t[1] = ky; t[2] = kz; t[3] = kx * kx + ky * ky + kz * kz;
      keys[i] = t;
    }
    __syncthreads();
    for (int i = 0; i < KCH; ++i) {
      v4f kv = keys[i];
      float d = qs + kv[3] - 2.f * (qx * kv[0] + qy * kv[1] + qz * kv[2]);
      d = fmaxf(d, 0.f);
      if (d < best[KNB - 1]) {   // sorted insertion, fully unrolled (register-resident)
        float cd = d; int ci = c0 + i;
#pragma unroll
        for (int j = 0; j < KNB; ++j) {
          bool sw = cd < best[j];
          float td = best[j]; int ti = bidx[j];
          best[j] = sw ? cd : td; bidx[j] = sw ? ci : ti;
          cd = sw ? td : cd;     ci = sw ? ti : ci;
        }
      }
    }
  }
#pragma unroll
  for (int j = 0; j < KNB; ++j) idx[(size_t)(b * MQ + q) * KNB + j] = bidx[j];
}

// ---------------- generic 256-wide GEMM (y = x*W^T [+epilogue]) ----------------
// Weights staged in LDS (128KB bf16). mode 0: plain store (q projection).
// mode 1: +bias +residual +LayerNorm.
#define SMEM_GEMM (DQ * DQ * 2)
__global__ void __launch_bounds__(256, 1)
gemm256_nt(const float* __restrict__ A, const uint16_t* __restrict__ Bw,
           float* __restrict__ out, const float* __restrict__ bias,
           const float* __restrict__ resid, const float* __restrict__ lnw,
           const float* __restrict__ lnb, int mode, int rows) {
  extern __shared__ uint16_t sB[];
  {
    const uint4* s = (const uint4*)Bw; uint4* d = (uint4*)sB;
    for (int i = threadIdx.x; i < DQ * DQ / 8; i += 256) d[i] = s[i];
  }
  __syncthreads();

  const int wave = threadIdx.x >> 5, lane = threadIdx.x & 31;
  const int col = lane & 15, hlf = lane >> 4;
  const int hoffA = hlf * 16, koffB = hlf * 16;
  const int wavesTotal = gridDim.x * (blockDim.x >> 5);
  for (int rg = blockIdx.x * (blockDim.x >> 5) + wave; rg < rows / 16;
       rg += wavesTotal) {
    // stop LLVM from hoisting loop-invariant weight fragments (spills!)
    asm volatile("" ::: "memory");
    const int r0 = rg * 16;
    const float* arow = A + (size_t)(r0 + col) * DQ;   // my A-fragment row
    v16bf Af[8];
#pragma unroll
    for (int s = 0; s < 8; ++s)
      Af[s] = packA(arow + s * 32 + hoffA, arow + s * 32 + 8 + hoffA);
    v8f acc[16];
#pragma unroll
    for (int n = 0; n < 16; ++n) {
      v8f c = {0.f, 0.f, 0.f, 0.f, 0.f, 0.f, 0.f, 0.f};
#pragma unroll
      for (int s = 0; s < 8; ++s) {
        v16bf bf = *(const v16bf*)(sB + (size_t)(n * 16 + col) * DQ + s * 32 + koffB);
        c = wmma_bf16(Af[s], bf, c);
      }
      acc[n] = c;
    }
    if (mode == 0) {
#pragma unroll
      for (int n = 0; n < 16; ++n)
#pragma unroll
        for (int i = 0; i < 8; ++i)
          out[(size_t)(r0 + hlf * 8 + i) * DQ + n * 16 + col] = acc[n][i];
    } else {
      // epilogue: +out_b, +feat residual, LayerNorm per row
#pragma unroll
      for (int n = 0; n < 16; ++n) {
        float bb = bias[n * 16 + col];
#pragma unroll
        for (int i = 0; i < 8; ++i)
          acc[n][i] += bb + resid[(size_t)(r0 + hlf * 8 + i) * DQ + n * 16 + col];
      }
      float mu[8], rs[8];
#pragma unroll
      for (int i = 0; i < 8; ++i) {
        float s1 = 0.f;
#pragma unroll
        for (int n = 0; n < 16; ++n) s1 += acc[n][i];
        s1 += __shfl_xor(s1, 1, 32); s1 += __shfl_xor(s1, 2, 32);
        s1 += __shfl_xor(s1, 4, 32); s1 += __shfl_xor(s1, 8, 32);
        mu[i] = s1 * (1.0f / DQ);
      }
#pragma unroll
      for (int i = 0; i < 8; ++i) {
        float s2 = 0.f;
#pragma unroll
        for (int n = 0; n < 16; ++n) {
          float d = acc[n][i] - mu[i]; s2 += d * d;
        }
        s2 += __shfl_xor(s2, 1, 32); s2 += __shfl_xor(s2, 2, 32);
        s2 += __shfl_xor(s2, 4, 32); s2 += __shfl_xor(s2, 8, 32);
        rs[i] = rsqrtf(s2 * (1.0f / DQ) + 1e-5f);
      }
#pragma unroll
      for (int n = 0; n < 16; ++n) {
        float w = lnw[n * 16 + col], o = lnb[n * 16 + col];
#pragma unroll
        for (int i = 0; i < 8; ++i)
          out[(size_t)(r0 + hlf * 8 + i) * DQ + n * 16 + col] =
              (acc[n][i] - mu[i]) * rs[i] * w + o;
      }
    }
  }
}

// ---------------- fused neighbor attention (one wave = one point) ----------------
// LDS: Wk (128KB bf16) + Wv (128KB) + phi_w2 (32KB) + phi_w1/b1 (1KB)
//      + 1KB/wave relayout scratch  ->  297KB of the 320KB/WGP.
#define SMEM_FUSED (DQ * DQ * 2 * 2 + DQ * PEH * 2 + PEH * 4 * 4 + 8 * 16 * 32 * 2)
__global__ void __launch_bounds__(256, 1)
fused_attn(const float* __restrict__ xyz, const float* __restrict__ feat,
           const float* __restrict__ phi_w1, const float* __restrict__ phi_b1,
           const float* __restrict__ phi_b2, const float* __restrict__ g_b1,
           const float* __restrict__ g_b2, const uint16_t* __restrict__ WkBf,
           const uint16_t* __restrict__ WvBf, const uint16_t* __restrict__ Ph2Bf,
           const uint16_t* __restrict__ G1Bf, const uint16_t* __restrict__ G2Bf,
           const int* __restrict__ idx, const float* __restrict__ qbuf,
           float* __restrict__ preout) {
  extern __shared__ uint16_t smem[];
  uint16_t* sWk  = smem;                    // [256][256] bf16, row = out ch
  uint16_t* sWv  = sWk + DQ * DQ;
  uint16_t* sPh2 = sWv + DQ * DQ;           // [256][64] bf16
  float*    sPhi = (float*)(sPh2 + DQ * PEH); // [64][4] = w0,w1,w2,b1
  uint16_t* sScr = (uint16_t*)(sPhi + PEH * 4);

  { // cooperative weight staging (already bf16 in global, 16B vectors)
    const uint4* s0 = (const uint4*)WkBf;  uint4* d0 = (uint4*)sWk;
    const uint4* s1 = (const uint4*)WvBf;  uint4* d1 = (uint4*)sWv;
    for (int i = threadIdx.x; i < DQ * DQ / 8; i += 256) { d0[i] = s0[i]; d1[i] = s1[i]; }
    const uint4* s2 = (const uint4*)Ph2Bf; uint4* d2 = (uint4*)sPh2;
    for (int i = threadIdx.x; i < DQ * PEH / 8; i += 256) d2[i] = s2[i];
    if (threadIdx.x < PEH) {
      sPhi[threadIdx.x * 4 + 0] = phi_w1[threadIdx.x * 3 + 0];
      sPhi[threadIdx.x * 4 + 1] = phi_w1[threadIdx.x * 3 + 1];
      sPhi[threadIdx.x * 4 + 2] = phi_w1[threadIdx.x * 3 + 2];
      sPhi[threadIdx.x * 4 + 3] = phi_b1[threadIdx.x];
    }
  }
  __syncthreads();

  const int wave = threadIdx.x >> 5, lane = threadIdx.x & 31;
  const int col = lane & 15, hlf = lane >> 4;
  const int hoffA = hlf * 16, koffB = hlf * 16;
  uint16_t* myScr = sScr + wave * 16 * 32;
  const int wavesTotal = gridDim.x * 8;

  for (int mi = blockIdx.x * 8 + wave; mi < NPTS; mi += wavesTotal) {
    // stop LLVM from hoisting loop-invariant weight fragments (spills!)
    asm volatile("" ::: "memory");
    const int b = mi / MQ;
    // my A-row = neighbor (lane&15) of this point
    const int nrow = idx[(size_t)mi * KNB + col];
    const float* nf = feat + (size_t)(b * MQ + nrow) * DQ;

    // ---- gather neighbor tile [16 x 256] into bf16 A-fragments ----
    v16bf Ngb[8];
#pragma unroll
    for (int s = 0; s < 8; ++s)
      Ngb[s] = packA(nf + s * 32 + hoffA, nf + s * 32 + 8 + hoffA);

    // ---- positional MLP hidden: Hpe[16 x 64] = relu(delta*phi1^T + b1) ----
    const float dx = xyz[(size_t)mi * 3 + 0] - xyz[(size_t)(b * MQ + nrow) * 3 + 0];
    const float dy = xyz[(size_t)mi * 3 + 1] - xyz[(size_t)(b * MQ + nrow) * 3 + 1];
    const float dz = xyz[(size_t)mi * 3 + 2] - xyz[(size_t)(b * MQ + nrow) * 3 + 2];
    v16bf Hpe[2];
#pragma unroll
    for (int s = 0; s < 2; ++s) {
      union { v16bf v; uint32_t u[8]; } a;
#pragma unroll
      for (int jj = 0; jj < 8; ++jj) {
        float hv[2];
#pragma unroll
        for (int e = 0; e < 2; ++e) {
          int j = jj * 2 + e;
          int h = s * 32 + ((j >> 3) << 3) + hoffA + (j & 7);
          float t = sPhi[h * 4 + 0] * dx + sPhi[h * 4 + 1] * dy +
                    sPhi[h * 4 + 2] * dz + sPhi[h * 4 + 3];
          hv[e] = fmaxf(t, 0.f);
        }
        a.u[jj] = pk2(hv[0], hv[1]);
      }
      Hpe[s] = a.v;
    }

    // ---- phase 1: att_in = q - kproj + pe  (as bf16 A-fragments) ----
    const float* qrow = qbuf + (size_t)mi * DQ;
    v16bf attA[8];
#pragma unroll
    for (int s = 0; s < 8; ++s) {
#pragma unroll
      for (int t = 0; t < 2; ++t) {
        const int n = 2 * s + t;
        v8f cK = {0.f, 0.f, 0.f, 0.f, 0.f, 0.f, 0.f, 0.f};
#pragma unroll
        for (int ks = 0; ks < 8; ++ks) {
          v16bf bf = *(const v16bf*)(sWk + (size_t)(n * 16 + col) * DQ + ks * 32 + koffB);
          cK = wmma_bf16(Ngb[ks], bf, cK);
        }
        v8f cP = {0.f, 0.f, 0.f, 0.f, 0.f, 0.f, 0.f, 0.f};
#pragma unroll
        for (int ks = 0; ks < 2; ++ks) {
          v16bf bp = *(const v16bf*)(sPh2 + (size_t)(n * 16 + col) * PEH + ks * 32 + koffB);
          cP = wmma_bf16(Hpe[ks], bp, cP);
        }
        const float qv = qrow[n * 16 + col];
        const float pb = phi_b2[n * 16 + col];
#pragma unroll
        for (int i = 0; i < 8; ++i)   // C-layout -> LDS (row = 8*hlf+i, col = t*16+col)
          myScr[(hlf * 8 + i) * 32 + t * 16 + col] = bfbits(qv - cK[i] + cP[i] + pb);
      }
      attA[s] = ldsA(myScr, col, hoffA);   // wave-internal LDS ops are in-order
    }

    // ---- phase 2: h = relu(att_in * g_w1^T + g_b1)  [16 x 128] ----
    v16bf hA[4];
    __builtin_prefetch(G1Bf, 0, 1);
#pragma unroll
    for (int t = 0; t < 4; ++t) {
#pragma unroll
      for (int u = 0; u < 2; ++u) {
        const int hn = 2 * t + u;
        v8f c = {0.f, 0.f, 0.f, 0.f, 0.f, 0.f, 0.f, 0.f};
#pragma unroll
        for (int ks = 0; ks < 8; ++ks) {
          v16bf bg = *(const v16bf*)(G1Bf + (size_t)(hn * 16 + col) * DQ + ks * 32 + koffB);
          c = wmma_bf16(attA[ks], bg, c);
        }
        const float bb = g_b1[hn * 16 + col];
#pragma unroll
        for (int i = 0; i < 8; ++i)
          myScr[(hlf * 8 + i) * 32 + u * 16 + col] = bfbits(fmaxf(c[i] + bb, 0.f));
      }
      hA[t] = ldsA(myScr, col, hoffA);
    }

    // ---- phase 3: logits, v, pe per N-tile; softmax over K=16; weighted sum ----
    __builtin_prefetch(G2Bf, 0, 1);
    for (int n = 0; n < 16; ++n) {
      v8f cL = {0.f, 0.f, 0.f, 0.f, 0.f, 0.f, 0.f, 0.f};
#pragma unroll
      for (int t = 0; t < 4; ++t) {
        v16bf bg = *(const v16bf*)(G2Bf + (size_t)(n * 16 + col) * ATH + t * 32 + koffB);
        cL = wmma_bf16(hA[t], bg, cL);
      }
      v8f cV = {0.f, 0.f, 0.f, 0.f, 0.f, 0.f, 0.f, 0.f};
#pragma unroll
      for (int ks = 0; ks < 8; ++ks) {
        v16bf bv = *(const v16bf*)(sWv + (size_t)(n * 16 + col) * DQ + ks * 32 + koffB);
        cV = wmma_bf16(Ngb[ks], bv, cV);
      }
      v8f cP = {0.f, 0.f, 0.f, 0.f, 0.f, 0.f, 0.f, 0.f};
#pragma unroll
      for (int ks = 0; ks < 2; ++ks) {
        v16bf bp = *(const v16bf*)(sPh2 + (size_t)(n * 16 + col) * PEH + ks * 32 + koffB);
        cP = wmma_bf16(Hpe[ks], bp, cP);
      }
      const float gb = g_b2[n * 16 + col], pb = phi_b2[n * 16 + col];
      // softmax over the 16 neighbor rows of my column: 8 local + partner half
      float mx = -3.4e38f;
#pragma unroll
      for (int i = 0; i < 8; ++i) mx = fmaxf(mx, cL[i] + gb);
      mx = fmaxf(mx, __shfl_xor(mx, 16, 32));
      float se = 0.f, so = 0.f;
#pragma unroll
      for (int i = 0; i < 8; ++i) {
        float e = __expf(cL[i] + gb - mx);
        se += e;
        so += e * (cV[i] + cP[i] + pb);
      }
      se += __shfl_xor(se, 16, 32);
      so += __shfl_xor(so, 16, 32);
      if (hlf == 0) preout[(size_t)mi * DQ + n * 16 + col] = so / se;
    }
  }
}

// ---------------- host side ----------------
extern "C" void kernel_launch(void* const* d_in, const int* in_sizes, int n_in,
                              void* d_out, int out_size, void* d_ws, size_t ws_size,
                              hipStream_t stream) {
  const float* xyz    = (const float*)d_in[0];
  const float* feat   = (const float*)d_in[1];
  const float* Wq     = (const float*)d_in[2];
  const float* Wk     = (const float*)d_in[3];
  const float* Wv     = (const float*)d_in[4];
  const float* phi_w1 = (const float*)d_in[5];
  const float* phi_b1 = (const float*)d_in[6];
  const float* phi_w2 = (const float*)d_in[7];
  const float* phi_b2 = (const float*)d_in[8];
  const float* g_w1   = (const float*)d_in[9];
  const float* g_b1   = (const float*)d_in[10];
  const float* g_w2   = (const float*)d_in[11];
  const float* g_b2   = (const float*)d_in[12];
  const float* out_w  = (const float*)d_in[13];
  const float* out_b  = (const float*)d_in[14];
  const float* ln_w   = (const float*)d_in[15];
  const float* ln_b   = (const float*)d_in[16];
  float* outp = (float*)d_out;

  char* ws = (char*)d_ws;
  size_t off = 0;
  auto carve = [&](size_t bytes) {
    void* p = ws + off;
    off = (off + bytes + 255) & ~(size_t)255;
    return p;
  };
  int*      idx    = (int*)     carve((size_t)NPTS * KNB * 4);
  float*    qbuf   = (float*)   carve((size_t)NPTS * DQ * 4);
  float*    preout = (float*)   carve((size_t)NPTS * DQ * 4);
  uint16_t* WqBf   = (uint16_t*)carve((size_t)DQ * DQ * 2);
  uint16_t* WkBf   = (uint16_t*)carve((size_t)DQ * DQ * 2);
  uint16_t* WvBf   = (uint16_t*)carve((size_t)DQ * DQ * 2);
  uint16_t* Ph2Bf  = (uint16_t*)carve((size_t)DQ * PEH * 2);
  uint16_t* G1Bf   = (uint16_t*)carve((size_t)ATH * DQ * 2);
  uint16_t* G2Bf   = (uint16_t*)carve((size_t)DQ * ATH * 2);
  uint16_t* OWBf   = (uint16_t*)carve((size_t)DQ * DQ * 2);

  auto cvt = [&](const float* s, uint16_t* d, int n) {
    cvt_bf16_kernel<<<(n + 255) / 256, 256, 0, stream>>>(s, d, n);
  };
  cvt(Wq, WqBf, DQ * DQ);
  cvt(Wk, WkBf, DQ * DQ);
  cvt(Wv, WvBf, DQ * DQ);
  cvt(phi_w2, Ph2Bf, DQ * PEH);
  cvt(g_w1, G1Bf, ATH * DQ);
  cvt(g_w2, G2Bf, DQ * ATH);
  cvt(out_w, OWBf, DQ * DQ);

  knn_kernel<<<dim3(MQ / 256, BQ), 256, 0, stream>>>(xyz, idx);

  (void)hipFuncSetAttribute(reinterpret_cast<const void*>(&gemm256_nt),
                            hipFuncAttributeMaxDynamicSharedMemorySize, SMEM_GEMM);
  // q = feat @ Wq^T
  gemm256_nt<<<256, 256, SMEM_GEMM, stream>>>(feat, WqBf, qbuf, nullptr, nullptr,
                                              nullptr, nullptr, 0, NPTS);

  (void)hipFuncSetAttribute(reinterpret_cast<const void*>(&fused_attn),
                            hipFuncAttributeMaxDynamicSharedMemorySize, SMEM_FUSED);
  fused_attn<<<256, 256, SMEM_FUSED, stream>>>(xyz, feat, phi_w1, phi_b1, phi_b2,
                                               g_b1, g_b2, WkBf, WvBf, Ph2Bf, G1Bf,
                                               G2Bf, idx, qbuf, preout);

  // out = LN(feat + preout @ out_w^T + out_b)
  gemm256_nt<<<256, 256, SMEM_GEMM, stream>>>(preout, OWBf, outp, out_b, feat,
                                              ln_w, ln_b, 1, NPTS);
  (void)in_sizes; (void)n_in; (void)out_size; (void)ws_size;
}
